// SwinMambaTex_6803228197398
// MI455X (gfx1250) — matrix operand order repaired and checked by reading, not compiled, not run on hardware
//
#include <hip/hip_runtime.h>
#include <math.h>

// ---------------- CDNA5 WMMA types ----------------
typedef __attribute__((ext_vector_type(16))) _Float16 v16h;
typedef __attribute__((ext_vector_type(8)))  float    v8f;

union Frag16 { v16h v; unsigned int u[8]; };

static constexpr int kT = 256, kS = 576;
static constexpr int kVocab = 8000;
static constexpr long long kBT = 1024;   // B*T rows
static constexpr long long kBS = 2304;   // B*S rows

__device__ __forceinline__ float sigmoidf_(float x) { return 1.0f / (1.0f + expf(-x)); }

// 16-byte async global->LDS copy (ASYNCcnt-tracked, CDNA5).
// lds_dst: generic pointer to __shared__ (low 32 bits == LDS byte address per ISA 10.2)
// sbase:   uniform global base (SGPR pair), voff: per-lane byte offset
__device__ __forceinline__ void async_cp16(void* lds_dst, const void* sbase, unsigned voff) {
#if defined(__AMDGCN__)
    unsigned lds = (unsigned)(unsigned long long)lds_dst;
    asm volatile("global_load_async_to_lds_b128 %0, %1, %2"
                 :
                 : "v"(lds), "v"(voff), "s"(sbase)
                 : "memory");
#else
    (void)lds_dst; (void)sbase; (void)voff;
#endif
}
__device__ __forceinline__ void async_wait0() {
#if defined(__AMDGCN__)
    asm volatile("s_wait_asynccnt 0x0" ::: "memory");
#endif
}
// wait until at most 4 async copies (the next tile's) remain in flight:
// previous tile complete (in-order completion), next tile still overlapping.
__device__ __forceinline__ void async_wait4() {
#if defined(__AMDGCN__)
    asm volatile("s_wait_asynccnt 0x4" ::: "memory");
#endif
}

// =====================================================================
//  f16 NT GEMM on v_wmma_f32_16x16x32_f16, double-buffered async pipeline:
//     C[z] = epi( A[z](MxK,f16) * B[z](NxK,f16)^T * alpha + bias )
//  Requirements (guaranteed by host orchestration):
//     M % 128 == 0, K % 32 == 0, B buffer padded+zeroed to ceil(N/128)*128 rows.
//  flags bit0: exact GELU; bit1: store f16 (C reinterpreted as _Float16*).
// =====================================================================
__global__ __launch_bounds__(256) void smx_gemm_h(
    const _Float16* __restrict__ A, const _Float16* __restrict__ Bm,
    float* C, const float* bias, const float* res,
    int M, int N, int K,
    long long sA, long long sB, long long sC,
    float alpha, int flags)
{
    __shared__ _Float16 As[2][128][32];   // 64B rows, 16B-aligned chunks
    __shared__ _Float16 Bs[2][128][32];   // stored [n][k]

    const _Float16* Ab = A  + (long long)blockIdx.z * sA;
    const _Float16* Bb = Bm + (long long)blockIdx.z * sB;
    float*          Cb = C  + (long long)blockIdx.z * sC;

    const int m0 = blockIdx.y * 128, n0 = blockIdx.x * 128;
    const int tid  = threadIdx.x;
    const int lane = tid & 31, wave = tid >> 5;
    const int wm = wave >> 1, wn = wave & 1;          // 4 x 2 wave grid
    const int hi = lane >> 4, li = lane & 15;

    v8f acc[2][4];
    {
        v8f zf = {};
        for (int i = 0; i < 2; ++i)
            for (int j = 0; j < 4; ++j) acc[i][j] = zf;
    }

    // stage one 128x32 K-slab of A and B into LDS buffer `buf` (4 asyncs/lane)
    auto stage = [&](int buf, int k0) {
        #pragma unroll
        for (int it = 0; it < 2; ++it) {
            int c  = tid + it * 256;
            int r  = c >> 2;               // tile row
            int pc = (c & 3) << 3;         // 16B piece, in halves
            unsigned avoff = (unsigned)((((long long)(m0 + r) * K) + k0 + pc) * 2);
            async_cp16(&As[buf][r][pc], Ab, avoff);
            unsigned bvoff = (unsigned)((((long long)(n0 + r) * K) + k0 + pc) * 2);
            async_cp16(&Bs[buf][r][pc], Bb, bvoff);
        }
    };

    const int NT = K >> 5;
    stage(0, 0);
    for (int i = 0; i < NT; ++i) {
        const int cur = i & 1;
        if (i + 1 < NT) {
            stage(1 - cur, (i + 1) << 5);   // DMA next tile while computing this one
            async_wait4();                  // tile i resident; tile i+1 in flight
        } else {
            async_wait0();
        }
        if (i + 2 < NT)                     // warm GL2 two slabs ahead
            __builtin_prefetch(Ab + (long long)(m0 + (tid & 127)) * K + ((i + 2) << 5), 0, 1);
        __syncthreads();

        // ---- build fragments per ISA 16-bit A(16x32)/B(32x16) layouts ----
        Frag16 af[2], bf[4];
        #pragma unroll
        for (int mt = 0; mt < 2; ++mt) {
            int mr = wm * 32 + mt * 16 + li;
            #pragma unroll
            for (int r = 0; r < 8; ++r) {
                int kk = ((r >> 2) << 4) + (hi << 3) + ((r & 3) << 1);
                af[mt].u[r] = *reinterpret_cast<const unsigned int*>(&As[cur][mr][kk]);
            }
        }
        #pragma unroll
        for (int nt = 0; nt < 4; ++nt) {
            int nc = wn * 64 + nt * 16 + li;
            #pragma unroll
            for (int r = 0; r < 8; ++r)
                bf[nt].u[r] = *reinterpret_cast<const unsigned int*>(&Bs[cur][nc][((r >> 2) << 4) + (hi << 3) + ((r & 3) << 1)]);
        }
        #pragma unroll
        for (int mt = 0; mt < 2; ++mt)
            #pragma unroll
            for (int nt = 0; nt < 4; ++nt)
                acc[mt][nt] = __builtin_amdgcn_wmma_f32_16x16x32_f16(
                    false, af[mt].v, false, bf[nt].v, (short)0, acc[mt][nt], false, false);
        __syncthreads();   // protects buf[cur] before iteration i+1 re-stages it
    }

    // ---- epilogue: D layout vgpr v -> row v + hi*8, col = li ----
    #pragma unroll
    for (int mt = 0; mt < 2; ++mt)
        #pragma unroll
        for (int nt = 0; nt < 4; ++nt) {
            int col = n0 + wn * 64 + nt * 16 + li;
            if (col >= N) continue;
            int rbase = m0 + wm * 32 + mt * 16 + hi * 8;
            float bv = bias ? bias[col] : 0.f;
            #pragma unroll
            for (int v = 0; v < 8; ++v) {
                int r = rbase + v;
                float f = acc[mt][nt][v] * alpha + bv;
                if (flags & 1) f = 0.5f * f * (1.0f + erff(f * 0.70710678118654752f));
                if (res) f += res[(long long)r * N + col];
                if (flags & 2) ((_Float16*)Cb)[(long long)r * N + col] = (_Float16)f;
                else           Cb[(long long)r * N + col] = f;
            }
        }
}

// =====================================================================
//  Conversion / element-wise kernels
// =====================================================================
__global__ void smx_fill(float* p, long long n) {
    long long i = (long long)blockIdx.x * 256 + threadIdx.x;
    if (i < n) p[i] = 0.f;
}
__global__ void smx_fillh(_Float16* p, long long n) {
    long long i = (long long)blockIdx.x * 256 + threadIdx.x;
    if (i < n) p[i] = (_Float16)0.f;
}
// straight f32 -> f16 with zero tail padding
__global__ void smx_h(const float* __restrict__ src, _Float16* dst, long long nsrc, long long ntot) {
    long long i = (long long)blockIdx.x * 256 + threadIdx.x;
    if (i >= ntot) return;
    dst[i] = (i < nsrc) ? (_Float16)src[i] : (_Float16)0.f;
}
// weight [K,N] f32 -> [Np,K] f16 (transpose + pad)
__global__ void smx_w2ht(const float* __restrict__ src, _Float16* dst, int K, int N, int Np) {
    long long i = (long long)blockIdx.x * 256 + threadIdx.x;
    if (i >= (long long)Np * K) return;
    int n = (int)(i / K), k = (int)(i % K);
    float v = (n < N) ? src[(long long)k * N + n] : 0.f;
    dst[i] = (_Float16)v;
}

__global__ void smx_embed(const int* tok, const float* __restrict__ emb, float* out, int rows) {
    long long i = (long long)blockIdx.x * 256 + threadIdx.x;
    if (i >= (long long)rows * 512) return;
    int r = (int)(i >> 9), c = (int)(i & 511);
    out[i] = emb[(long long)tok[r] * 512 + c] * 22.627416997969522f;  // sqrt(512)
}

__global__ __launch_bounds__(256) void smx_ln(const float* __restrict__ x,
                                              const float* __restrict__ g,
                                              const float* __restrict__ b,
                                              _Float16* out, int cols) {
    int row = blockIdx.x, tid = threadIdx.x;
    const float* p = x + (long long)row * cols;
    __shared__ float s1[256], s2[256];
    float a = 0.f, q = 0.f;
    for (int c = tid; c < cols; c += 256) { float v = p[c]; a += v; q += v * v; }
    s1[tid] = a; s2[tid] = q; __syncthreads();
    for (int off = 128; off; off >>= 1) {
        if (tid < off) { s1[tid] += s1[tid + off]; s2[tid] += s2[tid + off]; }
        __syncthreads();
    }
    float mu = s1[0] / cols;
    float var = s2[0] / cols - mu * mu;
    float rs = rsqrtf(var + 1e-5f);
    for (int c = tid; c < cols; c += 256)
        out[(long long)row * cols + c] = (_Float16)((p[c] - mu) * rs * g[c] + b[c]);
}

__global__ void smx_dt(const float* __restrict__ zx, const float* __restrict__ dtb, float* dt) {
    int i = blockIdx.x * 256 + threadIdx.x;
    if (i >= 1024 * 16) return;
    int r = i >> 4, h = i & 15;
    float v = zx[(long long)r * 2192 + 2176 + h] + dtb[h];
    dt[i] = (v > 20.f) ? v : log1pf(expf(v));
}

__global__ void smx_conv(const float* __restrict__ zx, const float* __restrict__ cw,
                         const float* __restrict__ cb, float* out) {
    int i = blockIdx.x * 256 + threadIdx.x;
    if (i >= 1024 * 1152) return;
    int r = i / 1152, c = i % 1152;
    int t = r & 255;
    float acc = cb[c];
    #pragma unroll
    for (int k = 0; k < 4; ++k) {
        int tt = t - 3 + k;
        if (tt >= 0) acc += zx[(long long)(r - t + tt) * 2192 + 1024 + c] * cw[c * 4 + k];
    }
    out[i] = acc * sigmoidf_(acc);
}

// selective scan: one workgroup per (b, head); 256 threads = 64 p x 4 n-groups
__global__ __launch_bounds__(256) void smx_scan(const float* __restrict__ conv,
                                                const float* __restrict__ dt,
                                                const float* __restrict__ A_log,
                                                const float* __restrict__ Dv, float* y) {
    int bh = blockIdx.x, b = bh >> 4, h = bh & 15;
    int tid = threadIdx.x, p = tid & 63, ng = tid >> 6;
    __shared__ float Bsh[64], Csh[64], red[4][64];
    float hs[16];
    #pragma unroll
    for (int i = 0; i < 16; ++i) hs[i] = 0.f;
    float Ah = -expf(A_log[h]);
    float Dh = Dv[h];
    for (int t = 0; t < kT; ++t) {
        const float* row = conv + (long long)(b * kT + t) * 1152;
        if (tid < 64) Bsh[tid] = row[1024 + tid];
        else if (tid < 128) Csh[tid - 64] = row[1088 + (tid - 64)];
        __syncthreads();
        float dtv = dt[(b * kT + t) * 16 + h];
        float dA  = expf(dtv * Ah);
        float xv  = row[h * 64 + p];
        float dtx = dtv * xv;
        float part = 0.f;
        #pragma unroll
        for (int i = 0; i < 16; ++i) {
            int n = ng * 16 + i;
            hs[i] = hs[i] * dA + dtx * Bsh[n];
            part += hs[i] * Csh[n];
        }
        red[ng][p] = part;
        __syncthreads();
        if (ng == 0)
            y[(long long)(b * kT + t) * 1024 + h * 64 + p] =
                red[0][p] + red[1][p] + red[2][p] + red[3][p] + Dh * xv;
        __syncthreads();
    }
}

__global__ __launch_bounds__(256) void smx_grms(const float* __restrict__ y,
                                                const float* __restrict__ zx,
                                                const float* __restrict__ nw, _Float16* out) {
    int row = blockIdx.x, tid = threadIdx.x;
    const float* yp = y + (long long)row * 1024;
    const float* zp = zx + (long long)row * 2192;   // z = first 1024 cols
    __shared__ float red[256];
    float vals[4], q = 0.f;
    #pragma unroll
    for (int i = 0; i < 4; ++i) {
        int c = tid + i * 256;
        float z = zp[c];
        float t = yp[c] * z * sigmoidf_(z);
        vals[i] = t; q += t * t;
    }
    red[tid] = q; __syncthreads();
    for (int off = 128; off; off >>= 1) {
        if (tid < off) red[tid] += red[tid + off];
        __syncthreads();
    }
    float rs = rsqrtf(red[0] / 1024.f + 1e-5f);
    #pragma unroll
    for (int i = 0; i < 4; ++i) {
        int c = tid + i * 256;
        out[(long long)row * 1024 + c] = (_Float16)(vals[i] * rs * nw[c]);
    }
}

// pack [B, L, srow] slice -> f16 [(b*8+h)][L(zs rows)][64] with optional RoPE
__global__ void smx_rope(const float* __restrict__ src, int srow, int scol,
                         _Float16* dst, int L, int dorope, int zs, long long total) {
    long long i = (long long)blockIdx.x * 256 + threadIdx.x;
    if (i >= total) return;
    int d = (int)(i & 63);
    long long r = i >> 6;
    int t = (int)(r % L);
    int h = (int)((r / L) & 7);
    int b = (int)(r / ((long long)L * 8));
    long long base = ((long long)b * L + t) * srow + scol + h * 64;
    float v = src[base + d];
    if (dorope) {
        int dm = d & 31;
        float inv = expf(-(float)dm * 0.28782313662425572f);   // 10000^(-2dm/64)
        float f = (float)t * inv;
        float other = (d < 32) ? -src[base + d + 32] : src[base + d - 32];
        v = v * cosf(f) + other * sinf(f);
    }
    dst[((long long)(b * 8 + h) * zs + t) * 64 + d] = (_Float16)v;
}

// V slice of kv -> f16 transposed [z][128 pad rows d][576 k] for NT GEMM
__global__ void smx_packvT(const float* __restrict__ kv, _Float16* vrt) {
    long long i = (long long)blockIdx.x * 256 + threadIdx.x;
    if (i >= 32LL * 576 * 64) return;
    int d = (int)(i & 63);
    int s = (int)((i >> 6) % 576);
    int z = (int)(i / (576 * 64));
    int b = z >> 3, h = z & 7;
    float v = kv[((long long)(b * 576 + s)) * 1024 + 512 + h * 64 + d];
    vrt[(long long)z * (128 * 576) + (long long)d * 576 + s] = (_Float16)v;
}

__global__ void smx_unpack(const float* __restrict__ az, _Float16* aoh) {
    long long i = (long long)blockIdx.x * 256 + threadIdx.x;
    if (i >= 32LL * 256 * 64) return;
    int d = (int)(i & 63);
    int t = (int)((i >> 6) & 255);
    int z = (int)(i >> 14);
    int b = z >> 3, h = z & 7;
    aoh[((long long)(b * 256 + t)) * 512 + h * 64 + d] = (_Float16)az[i];
}

__global__ __launch_bounds__(256) void smx_smax(float* sc, _Float16* ph, float* cov) {
    int row = blockIdx.x, tid = threadIdx.x;     // row in [0, B*8*T)
    float* p = sc + (long long)row * kS;
    __shared__ float red[256];
    float m = -1e30f;
    for (int s = tid; s < kS; s += 256) m = fmaxf(m, p[s]);
    red[tid] = m; __syncthreads();
    for (int off = 128; off; off >>= 1) {
        if (tid < off) red[tid] = fmaxf(red[tid], red[tid + off]);
        __syncthreads();
    }
    m = red[0]; __syncthreads();
    float sum = 0.f;
    for (int s = tid; s < kS; s += 256) { float e = expf(p[s] - m); p[s] = e; sum += e; }
    red[tid] = sum; __syncthreads();
    for (int off = 128; off; off >>= 1) {
        if (tid < off) red[tid] += red[tid + off];
        __syncthreads();
    }
    float inv = 1.f / red[0];
    int b = row >> 11;                            // / (8*256)
    for (int s = tid; s < kS; s += 256) {
        float w = p[s] * inv;
        ph[(long long)row * kS + s] = (_Float16)w;
        atomicAdd(&cov[b * kS + s], w * (1.f / 2048.f));  // mean over 8 heads, 256 t
    }
}

__global__ __launch_bounds__(256) void smx_argmax(const float* __restrict__ lg, int* tok) {
    int row = blockIdx.x, tid = threadIdx.x;
    const float* p = lg + (long long)row * kVocab;
    __shared__ float vred[256]; __shared__ int ired[256];
    float best = -1e30f; int bi = 0;
    for (int c = tid; c < kVocab; c += 256) {
        float v = p[c];
        if (v > best) { best = v; bi = c; }
    }
    vred[tid] = best; ired[tid] = bi; __syncthreads();
    for (int off = 128; off; off >>= 1) {
        if (tid < off) {
            if (vred[tid + off] > vred[tid] ||
                (vred[tid + off] == vred[tid] && ired[tid + off] < ired[tid])) {
                vred[tid] = vred[tid + off]; ired[tid] = ired[tid + off];
            }
        }
        __syncthreads();
    }
    if (tid == 0) tok[row] = ired[0];
}

__global__ void smx_rev(const float* __restrict__ in, float* out) {
    long long i = (long long)blockIdx.x * 256 + threadIdx.x;
    if (i >= kBT * 512) return;
    int c = (int)(i & 511);
    int r = (int)(i >> 9);
    int b = r >> 8, t = r & 255;
    out[i] = in[(((long long)(b * 256 + (255 - t))) << 9) + c];
}

__global__ void smx_cat(const float* __restrict__ f, const float* __restrict__ bw, _Float16* cat) {
    long long i = (long long)blockIdx.x * 256 + threadIdx.x;
    if (i >= kBT * 1024) return;
    int r = (int)(i >> 10), c = (int)(i & 1023);
    float v = (c < 512) ? f[(long long)r * 512 + c] : bw[(long long)r * 512 + (c - 512)];
    cat[i] = (_Float16)v;
}

// =====================================================================
//  Host-side orchestration
// =====================================================================
struct MambaP {
    const float *in_proj, *conv_w, *conv_b, *dt_bias, *A_log, *D, *norm_w, *out_proj;
};
struct BlockP {
    MambaP ssm;
    const float *q_w, *kv_w, *out_w, *out_b, *w1, *b1, *w2, *b2;
    const float *ln1_g, *ln1_b, *ln2_g, *ln2_b, *ln3_g, *ln3_b;
};

static inline int cdiv(long long a, long long b) { return (int)((a + b - 1) / b); }

static inline void gemm(hipStream_t st, const _Float16* A, const _Float16* Bm, float* C,
                        const float* bias, const float* res, int M, int N, int K,
                        float alpha, int flags,
                        int batch = 1, long long sA = 0, long long sB = 0, long long sC = 0) {
    dim3 g((N + 127) / 128, M / 128, batch), b(256);
    smx_gemm_h<<<g, b, 0, st>>>(A, Bm, C, bias, res, M, N, K, sA, sB, sC, alpha, flags);
}

// convert a [K,N] f32 weight to padded [ceil(N/128)*128, K] f16 in wbuf
static inline void w2ht(hipStream_t st, const float* w, _Float16* wbuf, int K, int N) {
    int Np = ((N + 127) / 128) * 128;
    long long tot = (long long)Np * K;
    smx_w2ht<<<cdiv(tot, 256), 256, 0, st>>>(w, wbuf, K, N, Np);
}

static inline void run_mamba(hipStream_t st, float* x, const float* lg, const float* lb,
                             const MambaP& p, _Float16* hbh, float* zx, float* dtb, float* cvb,
                             float* ysc, _Float16* gnh, _Float16* wbuf) {
    smx_ln<<<1024, 256, 0, st>>>(x, lg, lb, hbh, 512);
    w2ht(st, p.in_proj, wbuf, 512, 2192);
    gemm(st, hbh, wbuf, zx, nullptr, nullptr, 1024, 2192, 512, 1.f, 0);
    smx_dt<<<(1024 * 16 + 255) / 256, 256, 0, st>>>(zx, p.dt_bias, dtb);
    smx_conv<<<(1024 * 1152 + 255) / 256, 256, 0, st>>>(zx, p.conv_w, p.conv_b, cvb);
    smx_scan<<<64, 256, 0, st>>>(cvb, dtb, p.A_log, p.D, ysc);
    smx_grms<<<1024, 256, 0, st>>>(ysc, zx, p.norm_w, gnh);
    w2ht(st, p.out_proj, wbuf, 1024, 512);
    gemm(st, gnh, wbuf, x, nullptr, x, 1024, 512, 1024, 1.f, 0);
}

extern "C" void kernel_launch(void* const* d_in, const int* in_sizes, int n_in,
                              void* d_out, int out_size, void* d_ws, size_t ws_size,
                              hipStream_t stream) {
    (void)in_sizes; (void)n_in; (void)out_size; (void)ws_size;
    int cur = 0;
    auto F = [&]() { return (const float*)d_in[cur++]; };

    const float* memory = F();
    const int* tgt = (const int*)d_in[cur++];

    const float* dEmbed = F();
    BlockP blk[8];
    for (int i = 0; i < 8; ++i) {
        MambaP& m = blk[i].ssm;
        m.in_proj = F(); m.conv_w = F(); m.conv_b = F(); m.dt_bias = F();
        m.A_log = F(); m.D = F(); m.norm_w = F(); m.out_proj = F();
        blk[i].q_w = F(); blk[i].kv_w = F(); blk[i].out_w = F(); blk[i].out_b = F();
        blk[i].w1 = F(); blk[i].b1 = F(); blk[i].w2 = F(); blk[i].b2 = F();
        blk[i].ln1_g = F(); blk[i].ln1_b = F(); blk[i].ln2_g = F(); blk[i].ln2_b = F();
        blk[i].ln3_g = F(); blk[i].ln3_b = F();
    }
    const float* dNormG = F(); const float* dNormB = F();

    const float* rEmbed = F();
    MambaP rf[4], rb[4];
    for (int i = 0; i < 4; ++i) {
        rf[i].in_proj = F(); rf[i].conv_w = F(); rf[i].conv_b = F(); rf[i].dt_bias = F();
        rf[i].A_log = F(); rf[i].D = F(); rf[i].norm_w = F(); rf[i].out_proj = F();
    }
    for (int i = 0; i < 4; ++i) {
        rb[i].in_proj = F(); rb[i].conv_w = F(); rb[i].conv_b = F(); rb[i].dt_bias = F();
        rb[i].A_log = F(); rb[i].D = F(); rb[i].norm_w = F(); rb[i].out_proj = F();
    }
    const float *flnG[4], *flnB[4], *blnG[4], *blnB[4];
    for (int i = 0; i < 4; ++i) flnG[i] = F();
    for (int i = 0; i < 4; ++i) flnB[i] = F();
    for (int i = 0; i < 4; ++i) blnG[i] = F();
    for (int i = 0; i < 4; ++i) blnB[i] = F();
    const float* mergeW = F();
    const float* rNormG = F(); const float* rNormB = F();

    // ---------------- workspace layout ----------------
    float* wsp = (float*)d_ws;
    auto alloc  = [&](long long n) { float* p = wsp; wsp += n; return p; };
    auto allocH = [&](long long n) { _Float16* p = (_Float16*)wsp; wsp += (n + 1) / 2; return p; };

    float* x   = alloc(kBT * 512);
    float* zx  = alloc(kBT * 2192);
    float* dtb = alloc(kBT * 16);
    float* cvb = alloc(kBT * 1152);
    float* ysc = alloc(kBT * 1024);
    float* qb  = alloc(kBT * 512);
    float* kvb = alloc(kBS * 1024);
    float* sc  = alloc(32LL * 256 * 576);
    float* az  = alloc(32LL * 256 * 64);
    float* fwd = alloc(kBT * 512);
    float* bwd = alloc(kBT * 512);
    float* tmp = alloc(kBT * 512);
    float* mg  = alloc(kBT * 512);
    int*   tok = (int*)alloc(kBT);

    _Float16* hbh  = allocH(kBT * 512);
    _Float16* gnh  = allocH(kBT * 1024);
    _Float16* f1h  = allocH(kBT * 2048);
    _Float16* aoh  = allocH(kBT * 512);
    _Float16* qrh  = allocH(32LL * 256 * 64);
    _Float16* krh  = allocH(32LL * 640 * 64);    // N=576 padded to 640 rows/z
    _Float16* vrt  = allocH(32LL * 128 * 576);   // N=64  padded to 128 rows/z
    _Float16* sch  = allocH(32LL * 256 * 576);
    _Float16* cath = allocH(kBT * 1024);
    _Float16* memh = allocH(kBS * 512);
    _Float16* embD = allocH(8064LL * 512);       // vocab padded 8000 -> 8064
    _Float16* embR = allocH(8064LL * 512);
    _Float16* wbuf = allocH(2304LL * 512);       // largest padded weight (in_proj)

    float* dlog = (float*)d_out;                            // [1024, 8000]
    float* rlog = dlog + (long long)kBT * kVocab;           // [1024, 8000]
    float* cov  = rlog + (long long)kBT * kVocab;           // [4, 576]

    // ---------------- one-time per-launch prep ----------------
    smx_fill<<<cdiv(2304, 256), 256, 0, stream>>>(cov, 2304);
    smx_fillh<<<cdiv(32LL * 640 * 64, 256), 256, 0, stream>>>(krh, 32LL * 640 * 64);
    smx_fillh<<<cdiv(32LL * 128 * 576, 256), 256, 0, stream>>>(vrt, 32LL * 128 * 576);
    smx_h<<<cdiv(kBS * 512, 256), 256, 0, stream>>>(memory, memh, kBS * 512, kBS * 512);
    smx_h<<<cdiv(8064LL * 512, 256), 256, 0, stream>>>(dEmbed, embD, 8000LL * 512, 8064LL * 512);
    smx_h<<<cdiv(8064LL * 512, 256), 256, 0, stream>>>(rEmbed, embR, 8000LL * 512, 8064LL * 512);
    smx_embed<<<cdiv(kBT * 512, 256), 256, 0, stream>>>(tgt, dEmbed, x, 1024);

    // ---------------- draft blocks ----------------
    for (int i = 0; i < 8; ++i) {
        const BlockP& bp = blk[i];
        run_mamba(stream, x, bp.ln1_g, bp.ln1_b, bp.ssm, hbh, zx, dtb, cvb, ysc, gnh, wbuf);

        // cross attention
        smx_ln<<<1024, 256, 0, stream>>>(x, bp.ln2_g, bp.ln2_b, hbh, 512);
        w2ht(stream, bp.q_w, wbuf, 512, 512);
        gemm(stream, hbh, wbuf, qb, nullptr, nullptr, 1024, 512, 512, 1.f, 0);
        w2ht(stream, bp.kv_w, wbuf, 512, 1024);
        gemm(stream, memh, wbuf, kvb, nullptr, nullptr, 2304, 1024, 512, 1.f, 0);
        {
            long long tq = 32LL * 256 * 64, tk = 32LL * 576 * 64;
            smx_rope<<<cdiv(tq, 256), 256, 0, stream>>>(qb, 512, 0, qrh, 256, 1, 256, tq);
            smx_rope<<<cdiv(tk, 256), 256, 0, stream>>>(kvb, 1024, 0, krh, 576, 1, 640, tk);
            smx_packvT<<<cdiv(tk, 256), 256, 0, stream>>>(kvb, vrt);
        }
        gemm(stream, qrh, krh, sc, nullptr, nullptr, 256, 576, 64, 0.125f, 0,
             32, 256LL * 64, 640LL * 64, 256LL * 576);
        smx_smax<<<8192, 256, 0, stream>>>(sc, sch, cov);
        gemm(stream, sch, vrt, az, nullptr, nullptr, 256, 64, 576, 1.f, 0,
             32, 256LL * 576, 128LL * 576, 256LL * 64);
        smx_unpack<<<cdiv(32LL * 256 * 64, 256), 256, 0, stream>>>(az, aoh);
        w2ht(stream, bp.out_w, wbuf, 512, 512);
        gemm(stream, aoh, wbuf, x, bp.out_b, x, 1024, 512, 512, 1.f, 0);

        // FFN: GELU fused, f16 intermediate
        smx_ln<<<1024, 256, 0, stream>>>(x, bp.ln3_g, bp.ln3_b, hbh, 512);
        w2ht(stream, bp.w1, wbuf, 512, 2048);
        gemm(stream, hbh, wbuf, (float*)f1h, bp.b1, nullptr, 1024, 2048, 512, 1.f, 1 | 2);
        w2ht(stream, bp.w2, wbuf, 2048, 512);
        gemm(stream, f1h, wbuf, x, bp.b2, x, 1024, 512, 2048, 1.f, 0);
    }

    // ---------------- draft logits + argmax ----------------
    smx_ln<<<1024, 256, 0, stream>>>(x, dNormG, dNormB, hbh, 512);
    gemm(stream, hbh, embD, dlog, nullptr, nullptr, 1024, 8000, 512, 1.f, 0);
    smx_argmax<<<1024, 256, 0, stream>>>(dlog, tok);

    // ---------------- refine ----------------
    smx_embed<<<cdiv(kBT * 512, 256), 256, 0, stream>>>(tok, rEmbed, fwd, 1024);
    smx_rev<<<cdiv(kBT * 512, 256), 256, 0, stream>>>(fwd, bwd);
    for (int i = 0; i < 4; ++i) {
        run_mamba(stream, fwd, flnG[i], flnB[i], rf[i], hbh, zx, dtb, cvb, ysc, gnh, wbuf);
        run_mamba(stream, bwd, blnG[i], blnB[i], rb[i], hbh, zx, dtb, cvb, ysc, gnh, wbuf);
    }
    smx_rev<<<cdiv(kBT * 512, 256), 256, 0, stream>>>(bwd, tmp);
    smx_cat<<<cdiv(kBT * 1024, 256), 256, 0, stream>>>(fwd, tmp, cath);
    w2ht(stream, mergeW, wbuf, 1024, 512);
    gemm(stream, cath, wbuf, mg, nullptr, nullptr, 1024, 512, 1024, 1.f, 0);
    smx_ln<<<1024, 256, 0, stream>>>(mg, rNormG, rNormB, hbh, 512);
    gemm(stream, hbh, embR, rlog, nullptr, nullptr, 1024, 8000, 512, 1.f, 0);
}
